// GRU_43396349559297
// MI455X (gfx1250) — matrix operand order repaired and checked by reading, not compiled, not run on hardware
//
#include <hip/hip_runtime.h>
#include <hip/hip_bf16.h>
#include <math.h>

// ---------------------------------------------------------------------------
// GRU for MI455X (gfx1250, wave32).
//   Phase 1: x_proj GEMM (M=32768,K=1024,N=3072) with v_wmma_f32_16x16x32_bf16;
//            operands stream from L2 (192MB holds inputs f32 + weights).
//   Phase 2: persistent recurrence kernel, 32 blocks, software grid barrier per
//            timestep. h (64x1024 bf16 = 128KB) is staged into LDS once per
//            step via GLOBAL_LOAD_ASYNC_TO_LDS_B128 (ASYNCcnt path); A frags
//            then come from ds_load_b128 with 6x reuse, cutting L2 traffic on
//            the serial critical path from ~12.6GB to ~2.1GB total.
// Precision: bf16 WMMA (16x16x32 = 8x MACs/instr vs f32 WMMA 16x16x4), f32 acc.
// ---------------------------------------------------------------------------

typedef __attribute__((ext_vector_type(16))) __bf16 v16bf;
typedef __attribute__((ext_vector_type(8)))  __bf16 v8bf;
typedef __attribute__((ext_vector_type(8)))  float  v8f;
typedef int v4i __attribute__((vector_size(16)));   // matches builtin pointee

#define BATCH   64
#define SEQ     512
#define IDIM    1024
#define HID     1024
#define GATES   3072          // 3*HID
#define M1      (BATCH*SEQ)   // 32768
#define NB_REC  32            // persistent blocks (32 * 32 cols = 1024)

#define SMEM_H_BYTES   (BATCH * HID * 2)                  // 128KB bf16 h tile
#define SMEM_GATES_F   (3 * 64 * 33)                      // padded gate buffer
#define SMEM_TOTAL     (SMEM_H_BYTES + SMEM_GATES_F * 4)  // ~153KB of 320KB LDS

#if __has_builtin(__builtin_amdgcn_global_load_async_to_lds_b128)
#define HAVE_ASYNC_LDS 1
#endif

// ----- fragment helpers (contiguous k = 16*hi + s, same in A and B) ----------------------

__device__ inline void cvt8(const float* __restrict__ p, v16bf& d, int base) {
#pragma unroll
  for (int i = 0; i < 8; ++i) d[base + i] = (__bf16)p[i];
}

// A frag (16x32 MxK, bf16): lane(l2,hi) = row l2; K = k0 + hi*16 .. +15 (contiguous)
__device__ inline v16bf load_a_f32(const float* __restrict__ rowp, int k0, int hi) {
  v16bf a;
  cvt8(rowp + k0 + hi * 16,     a, 0);
  cvt8(rowp + k0 + hi * 16 + 8, a, 8);
  return a;
}
// B frag (32x16 KxN, bf16): lane(l2,hi) = col l2; K = k0 + hi*16 .. +15 (contiguous)
__device__ inline v16bf load_b_f32(const float* __restrict__ rowp, int k0, int hi) {
  v16bf b;
  const float* p = rowp + k0 + hi * 16;
  cvt8(p,     b, 0);
  cvt8(p + 8, b, 8);
  return b;
}
__device__ inline v16bf load_a_bf16(const __bf16* rowp, int k0, int hi) {
  v8bf lo  = *(const v8bf*)(rowp + k0 + hi * 16);
  v8bf hi8 = *(const v8bf*)(rowp + k0 + hi * 16 + 8);
  return __builtin_shufflevector(lo, hi8, 0,1,2,3,4,5,6,7,8,9,10,11,12,13,14,15);
}
__device__ inline v16bf load_b_bf16(const __bf16* __restrict__ rowp, int k0, int hi) {
  return *(const v16bf*)(rowp + k0 + hi * 16);
}

__device__ inline v8f wmma_bf16(v16bf a, v16bf b, v8f c) {
  return __builtin_amdgcn_wmma_f32_16x16x32_bf16(false, a, false, b, (short)0, c,
                                                 false, false);
}

__device__ inline v8f zero8() {
  v8f z = {0.f, 0.f, 0.f, 0.f, 0.f, 0.f, 0.f, 0.f};
  return z;
}

// ----- async global->LDS staging (CDNA5 ASYNCcnt path) ----------------------

__device__ inline void async_cp16(const void* gsrc, void* lds_dst) {
#ifdef HAVE_ASYNC_LDS
  __builtin_amdgcn_global_load_async_to_lds_b128(
      (__attribute__((address_space(1))) v4i*)gsrc,
      (__attribute__((address_space(3))) v4i*)lds_dst, 0, 0);
#else
  *(v4i*)lds_dst = *(const v4i*)gsrc;   // fallback: load + ds_store
#endif
}

__device__ inline void async_wait0() {
#ifdef HAVE_ASYNC_LDS
#if __has_builtin(__builtin_amdgcn_s_wait_asynccnt)
  __builtin_amdgcn_s_wait_asynccnt(0);
#else
  asm volatile("s_wait_asynccnt 0" ::: "memory");
#endif
#endif
}

// ----- software grid barrier (persistent kernel, NB_REC co-resident blocks) -

__device__ inline void grid_barrier(unsigned* __restrict__ bar, unsigned nb) {
  __syncthreads();
  if (threadIdx.x == 0) {
    __threadfence();
    unsigned gen = __atomic_load_n(bar + 1, __ATOMIC_ACQUIRE);
    if (atomicAdd(bar, 1u) == nb - 1u) {
      __atomic_store_n(bar, 0u, __ATOMIC_RELAXED);
      __atomic_fetch_add(bar + 1, 1u, __ATOMIC_RELEASE);
    } else {
      while (__atomic_load_n(bar + 1, __ATOMIC_ACQUIRE) == gen)
        __builtin_amdgcn_s_sleep(2);
    }
    __threadfence();
  }
  __syncthreads();
}

// ----- setup kernels --------------------------------------------------------

__global__ void f32_to_bf16_kernel(const float* __restrict__ s,
                                   __bf16* __restrict__ d, int n) {
  int i = blockIdx.x * blockDim.x + threadIdx.x;
  if (i < n) d[i] = (__bf16)s[i];
}

__global__ void init_h_kernel(const float* __restrict__ h0,
                              float* __restrict__ hF, __bf16* __restrict__ hB,
                              unsigned* __restrict__ bar) {
  int i = blockIdx.x * blockDim.x + threadIdx.x;
  if (i < BATCH * HID) {
    float v = h0[i];
    hF[i] = v;
    hB[i] = (__bf16)v;
  }
  if (i == 0) { bar[0] = 0u; bar[1] = 0u; }
}

// ----- phase 1: x_proj = inputs @ Wih^T + bias_ih ---------------------------
// block = 256 thr (8 waves), tile 128x128, wave tile 64x32 (4x2 WMMA accums).

__global__ void __launch_bounds__(256)
xproj_gemm_kernel(const float* __restrict__ X,    // (M1, IDIM)
                  const float* __restrict__ Wih,  // (GATES, IDIM)
                  const float* __restrict__ bih,  // (GATES)
                  float* __restrict__ xproj) {    // (M1, GATES)
  const int tid  = threadIdx.x;
  const int lane = tid & 31, l2 = lane & 15, hi = lane >> 4;
  const int wave = tid >> 5;
  const int wm = wave >> 2, wn = wave & 3;
  const int m0 = blockIdx.y * 128 + wm * 64;
  const int n0 = blockIdx.x * 128 + wn * 32;

  v8f acc[4][2];
#pragma unroll
  for (int mi = 0; mi < 4; ++mi)
#pragma unroll
    for (int ni = 0; ni < 2; ++ni) acc[mi][ni] = zero8();

  for (int k0 = 0; k0 < IDIM; k0 += 32) {
    v16bf a[4], b[2];
#pragma unroll
    for (int mi = 0; mi < 4; ++mi)
      a[mi] = load_a_f32(X + (size_t)(m0 + mi * 16 + l2) * IDIM, k0, hi);
#pragma unroll
    for (int ni = 0; ni < 2; ++ni)
      b[ni] = load_b_f32(Wih + (size_t)(n0 + ni * 16 + l2) * IDIM, k0, hi);
#pragma unroll
    for (int mi = 0; mi < 4; ++mi)
#pragma unroll
      for (int ni = 0; ni < 2; ++ni)
        acc[mi][ni] = wmma_bf16(a[mi], b[ni], acc[mi][ni]);
  }

#pragma unroll
  for (int ni = 0; ni < 2; ++ni) {
    const int col  = n0 + ni * 16 + l2;
    const float bv = bih[col];
#pragma unroll
    for (int mi = 0; mi < 4; ++mi)
#pragma unroll
      for (int v = 0; v < 8; ++v) {
        int row = m0 + mi * 16 + v + 8 * hi;   // C/D layout: M = v + 8*hi, N = l2
        xproj[(size_t)row * GATES + col] = acc[mi][ni][v] + bv;
      }
  }
}

// ----- phase 2: persistent recurrence --------------------------------------
// 32 blocks x 256 thr. Block owns 32 hidden columns; per step: async-stage h
// into LDS, compute the 3 gate GEMM slices (64x32 each, K=1024) with 24 WMMA
// tiles = 3 per wave (A frags from LDS, B frags from L2-resident Whh),
// exchange partials through LDS, gate math, write h_new, grid-barrier.

__global__ void __launch_bounds__(256)
gru_recurrence_kernel(const float*  __restrict__ xproj,  // (M1, GATES), bias_ih folded in
                      const __bf16* __restrict__ WhhB,   // (GATES, HID) bf16
                      const float*  __restrict__ bhh,    // (GATES)
                      float*        __restrict__ hF,     // (2, BATCH, HID) f32 ping-pong
                      __bf16*       __restrict__ hB,     // (2, BATCH, HID) bf16 ping-pong
                      float*        __restrict__ out,    // (BATCH,SEQ,HID) + (BATCH,HID)
                      unsigned*     __restrict__ bar) {
  extern __shared__ char smem[];
  __bf16* hS    = (__bf16*)smem;                       // 64x1024 bf16 (128KB)
  float*  gateS = (float*)(smem + SMEM_H_BYTES);       // [3][64][33]

  const int tid  = threadIdx.x;
  const int lane = tid & 31, l2 = lane & 15, hi = lane >> 4;
  const int wave = tid >> 5;
  const int c0   = blockIdx.x * 32;                    // hidden-column slice

  int gI[3], mI[3], nI[3];
#pragma unroll
  for (int ti = 0; ti < 3; ++ti) {
    int tIdx = wave * 3 + ti;                          // 0..23
    gI[ti] = tIdx >> 3;                                // gate 0..2
    mI[ti] = (tIdx & 7) >> 1;                          // batch tile 0..3
    nI[ti] = tIdx & 1;                                 // col tile 0..1
  }

  int cur = 0;
  for (int t = 0; t < SEQ; ++t) {
    // --- cooperative async stage: hB[cur] (128KB) -> LDS -------------------
    {
      const char* src = (const char*)(hB + (size_t)cur * BATCH * HID);
      char* dst = (char*)hS;
      // 8192 x 16B chunks / 256 threads = 32 per thread (uniform trip count,
      // EXEC all ones). Tracked by ASYNCcnt; each wave waits, then barrier.
#pragma unroll 4
      for (int i = tid; i < SMEM_H_BYTES / 16; i += 256)
        async_cp16(src + (size_t)i * 16, dst + (size_t)i * 16);
      async_wait0();
      __syncthreads();
    }

    v8f acc[3];
#pragma unroll
    for (int ti = 0; ti < 3; ++ti) acc[ti] = zero8();

    for (int k0 = 0; k0 < HID; k0 += 32) {
#pragma unroll
      for (int ti = 0; ti < 3; ++ti) {
        v16bf a = load_a_bf16(hS + (size_t)(mI[ti] * 16 + l2) * HID, k0, hi);
        v16bf b = load_b_bf16(
            WhhB + (size_t)(gI[ti] * HID + c0 + nI[ti] * 16 + l2) * HID, k0, hi);
        acc[ti] = wmma_bf16(a, b, acc[ti]);
      }
    }

    __syncthreads();   // all ds reads of hS done before gate buffer writes land
#pragma unroll
    for (int ti = 0; ti < 3; ++ti)
#pragma unroll
      for (int v = 0; v < 8; ++v)
        gateS[(gI[ti] * 64 + mI[ti] * 16 + v + 8 * hi) * 33 + nI[ti] * 16 + l2] =
            acc[ti][v];
    __syncthreads();

    const int nxt = cur ^ 1;
#pragma unroll
    for (int e = 0; e < 8; ++e) {
      int linear = e * 256 + tid;                      // 0..2047
      int b = linear >> 5;                             // batch 0..63
      int c = linear & 31;                             // col in slice
      int j = c0 + c;
      const float* xrow = xproj + ((size_t)b * SEQ + t) * GATES;
      float xr = xrow[j], xz = xrow[HID + j], xn = xrow[2 * HID + j];
      float aR = gateS[(0 * 64 + b) * 33 + c];
      float aZ = gateS[(1 * 64 + b) * 33 + c];
      float aN = gateS[(2 * 64 + b) * 33 + c];
      float r = 1.f / (1.f + expf(-(xr + aR + bhh[j])));
      float z = 1.f / (1.f + expf(-(xz + aZ + bhh[HID + j])));
      float n = tanhf(xn + r * (aN + bhh[2 * HID + j]));
      float hp = hF[(size_t)cur * BATCH * HID + b * HID + j];
      float hv = (1.f - z) * n + z * hp;
      out[((size_t)b * SEQ + t) * HID + j] = hv;
      hF[(size_t)nxt * BATCH * HID + b * HID + j] = hv;
      hB[(size_t)nxt * BATCH * HID + b * HID + j] = (__bf16)hv;
      if (t == SEQ - 1)
        out[(size_t)BATCH * SEQ * HID + b * HID + j] = hv;   // hn tail
    }

    grid_barrier(bar, gridDim.x);
    cur ^= 1;
  }
}

// ----- host side ------------------------------------------------------------

extern "C" void kernel_launch(void* const* d_in, const int* in_sizes, int n_in,
                              void* d_out, int out_size, void* d_ws, size_t ws_size,
                              hipStream_t stream) {
  const float* inputs  = (const float*)d_in[0];  // (64,512,1024)
  const float* h0      = (const float*)d_in[1];  // (1,64,1024)
  const float* Wih     = (const float*)d_in[2];  // (3072,1024)
  const float* Whh     = (const float*)d_in[3];  // (3072,1024)
  const float* bih     = (const float*)d_in[4];  // (3072)
  const float* bhh     = (const float*)d_in[5];  // (3072)
  float* out = (float*)d_out;

  // Workspace layout (bytes): barrier | Whh_bf16 | hF pingpong | hB pingpong | xproj
  char* ws = (char*)d_ws;
  size_t off = 0;
  unsigned* bar  = (unsigned*)(ws + off); off += 256;
  __bf16* WhhB   = (__bf16*)(ws + off);   off += (size_t)GATES * HID * 2;
  float*  hF     = (float*)(ws + off);    off += (size_t)2 * BATCH * HID * 4;
  __bf16* hB     = (__bf16*)(ws + off);   off += (size_t)2 * BATCH * HID * 2;
  float*  xproj  = (float*)(ws + off);    off += (size_t)M1 * GATES * 4;
  (void)ws_size; (void)out_size; (void)n_in; (void)in_sizes;

  // 1) bf16 copy of recurrent weights
  {
    int n = GATES * HID;
    f32_to_bf16_kernel<<<(n + 255) / 256, 256, 0, stream>>>(Whh, WhhB, n);
  }
  // 2) init h ping-pong + grid barrier state (every call: replay-safe)
  init_h_kernel<<<(BATCH * HID + 255) / 256, 256, 0, stream>>>(h0, hF, hB, bar);
  // 3) x_proj GEMM: grid (3072/128, 32768/128)
  {
    dim3 grid(GATES / 128, M1 / 128);
    xproj_gemm_kernel<<<grid, 256, 0, stream>>>(inputs, Wih, bih, xproj);
  }
  // 4) persistent recurrence: dynamic LDS = 128KB h tile + gate buffer
  gru_recurrence_kernel<<<NB_REC, 256, SMEM_TOTAL, stream>>>(
      xproj, WhhB, bhh, hF, hB, out, bar);
}
